// FastformerEncoderLayer_80779744903691
// MI455X (gfx1250) — matrix operand order repaired and checked by reading, not compiled, not run on hardware
//
#include <hip/hip_runtime.h>
#include <math.h>

// ---------------- CDNA5 WMMA types ----------------
typedef __attribute__((ext_vector_type(16))) __bf16 v16bf;
typedef __attribute__((ext_vector_type(8)))  __bf16 v8bf;
typedef __attribute__((ext_vector_type(8)))  float  v8f;

#define DMODEL 1024
#define MTOT   8192      // B*N = 8*1024 rows
#define HEADS  16
#define HD     64
#define FFDIM  4096

__device__ __forceinline__ unsigned short f32_to_bf16(float f) {
  unsigned int u = __float_as_uint(f);
  u += 0x7FFFu + ((u >> 16) & 1u);   // round-to-nearest-even
  return (unsigned short)(u >> 16);
}

// LDS byte offset (relative to wave LDS base) for async-to-LDS VDST operand
__device__ __forceinline__ unsigned lds_off(const void* p) {
  return (unsigned)(unsigned long long)
      (const __attribute__((address_space(3))) void*)p;
}

// one 16-byte global -> LDS DMA, tracked by ASYNCcnt
__device__ __forceinline__ void async_b128(unsigned ldsByteOff, const void* gaddr) {
  asm volatile("global_load_async_to_lds_b128 %0, %1, off"
               :: "v"(ldsByteOff), "v"((unsigned long long)gaddr)
               : "memory");
}

// ---------------- elementwise f32 -> bf16 ----------------
__global__ void cvt_bf16(const float* __restrict__ in, unsigned short* __restrict__ out, long n) {
  long i = (long)blockIdx.x * blockDim.x + threadIdx.x;
  long stride = (long)gridDim.x * blockDim.x;
  for (; i < n; i += stride) out[i] = f32_to_bf16(in[i]);
}

// ---------------- LayerNorm over last dim (1024), bf16 output ----------------
__global__ __launch_bounds__(256)
void ln_bf16(const float* __restrict__ x, const float* __restrict__ g,
             const float* __restrict__ b, unsigned short* __restrict__ y) {
  __shared__ float s1[256], s2[256];
  int row = blockIdx.x, tid = threadIdx.x;
  const float* xr = x + (long)row * DMODEL;
  float s = 0.f, q = 0.f;
  for (int i = tid; i < DMODEL; i += 256) { float v = xr[i]; s += v; q += v * v; }
  s1[tid] = s; s2[tid] = q; __syncthreads();
  for (int off = 128; off > 0; off >>= 1) {
    if (tid < off) { s1[tid] += s1[tid + off]; s2[tid] += s2[tid + off]; }
    __syncthreads();
  }
  float mean = s1[0] * (1.f / DMODEL);
  float var  = s2[0] * (1.f / DMODEL) - mean * mean;
  float rs = rsqrtf(var + 1e-5f);
  unsigned short* yr = y + (long)row * DMODEL;
  for (int i = tid; i < DMODEL; i += 256)
    yr[i] = f32_to_bf16((xr[i] - mean) * rs * g[i] + b[i]);
}

// ---------------- Fastformer global-key attention ----------------
// kv: [8192, 2048] f32 (cols 0..1023 = K proj, 1024..2047 = V proj)
__global__ __launch_bounds__(256)
void fast_attn(const float* __restrict__ kv, const float* __restrict__ key_w,
               const unsigned char* __restrict__ mask, unsigned short* __restrict__ attn_out) {
  __shared__ float sm[256], sp[256], mx[64], gk[64];
  int b = blockIdx.x >> 4, h = blockIdx.x & 15;
  int tid = threadIdx.x, d = tid & 63, c = tid >> 6;
  const float invScale = 0.125f;   // 1/sqrt(64)
  const float* kcol = kv + ((long)b * 1024) * 2048 + h * 64 + d;
  const float* kwr = key_w + h * 1024;
  const unsigned char* mr = mask + b * 1024;
  int nStart = c * 256, nEnd = nStart + 256;
  float m = -3.0e38f;
  for (int n = nStart; n < nEnd; ++n) {
    float kvv = kcol[(long)n * 2048];
    float a = mr[n] ? -1.0e9f : kvv * kwr[n] * invScale;
    m = fmaxf(m, a);
  }
  sm[tid] = m; __syncthreads();
  if (c == 0) mx[d] = fmaxf(fmaxf(sm[d], sm[64 + d]), fmaxf(sm[128 + d], sm[192 + d]));
  __syncthreads();
  float mm = mx[d];
  float s = 0.f, p = 0.f;
  for (int n = nStart; n < nEnd; ++n) {
    float kvv = kcol[(long)n * 2048];
    float a = mr[n] ? -1.0e9f : kvv * kwr[n] * invScale;
    float e = __expf(a - mm);
    s += e; p += e * kvv;
  }
  sm[tid] = s; sp[tid] = p; __syncthreads();
  if (c == 0) {
    float S = sm[d] + sm[64 + d] + sm[128 + d] + sm[192 + d];
    float P = sp[d] + sp[64 + d] + sp[128 + d] + sp[192 + d];
    gk[d] = P / S;
  }
  __syncthreads();
  const float* vbase = kv + ((long)b * 1024) * 2048 + 1024 + h * 64;
  unsigned short* obase = attn_out + ((long)b * 1024) * 1024 + h * 64;
  for (int idx = tid; idx < 1024 * 64; idx += 256) {
    int n = idx >> 6, dd = idx & 63;
    float v = vbase[(long)n * 2048 + dd];
    obase[(long)n * 1024 + dd] = f32_to_bf16(v * gk[dd]);
  }
}

// ---------------- gated ReLU: a * relu(g), bf16 output ----------------
__global__ void gate_bf16(const float* __restrict__ ff, unsigned short* __restrict__ gated) {
  long total = (long)MTOT * 2048;
  long i = (long)blockIdx.x * blockDim.x + threadIdx.x;
  long stride = (long)gridDim.x * blockDim.x;
  for (; i < total; i += stride) {
    long m = i >> 11, j = i & 2047;
    float a = ff[m * 4096 + j];
    float g = ff[m * 4096 + 2048 + j];
    gated[i] = f32_to_bf16(a * fmaxf(g, 0.f));
  }
}

// ---------------- WMMA GEMM: C[M,N] = A[M,K] * W[N,K]^T + bias (+resid) ----------------
// Double-buffered LDS tiles filled by GLOBAL_LOAD_ASYNC_TO_LDS_B128 DMA (ASYNCcnt),
// overlapped with v_wmma_f32_16x16x32_bf16 on the other buffer.
#define BM 128
#define BN 128
#define BK 32
#define LDT 40   // LDS row stride (bf16 elems), padded; row stride = 80B (16B-aligned)

template <bool RES>
__global__ __launch_bounds__(256)
void wmma_gemm_bf16(const unsigned short* __restrict__ A,   // [M,K] bf16 bits
                    const unsigned short* __restrict__ W,   // [N,K] bf16 bits
                    const float* __restrict__ bias,         // [N]
                    const float* __restrict__ resid,        // [M,N] (only if RES)
                    float* __restrict__ C,                  // [M,N]
                    int M, int N, int K) {
  __shared__ __align__(16) unsigned short ldsA[2][BM * LDT];
  __shared__ __align__(16) unsigned short ldsB[2][BN * LDT];
  int tid = threadIdx.x;
  int lane = tid & 31, wave = tid >> 5;
  int wm = (wave >> 2) * 64;       // wave M origin: 0 / 64
  int wn = (wave & 3) * 32;        // wave N origin: 0/32/64/96
  int l = lane & 15, hi = lane >> 4;
  long m0 = (long)blockIdx.y * BM;
  long n0 = (long)blockIdx.x * BN;

  // per-thread 16B staging chunks: tile is 128 rows x 32 cols = 512 chunks / matrix
  int r0 = tid >> 2;                 // rows 0..63
  int r1 = r0 + 64;                  // rows 64..127
  int kc = (tid & 3) * 8;            // col 0/8/16/24
  const unsigned short* Abase = A + m0 * K + kc;
  const unsigned short* Wbase = W + n0 * K + kc;

  v8f zero = {0.f, 0.f, 0.f, 0.f, 0.f, 0.f, 0.f, 0.f};
  v8f acc[4][2];
#pragma unroll
  for (int i = 0; i < 4; ++i)
#pragma unroll
    for (int j = 0; j < 2; ++j) acc[i][j] = zero;

  // ---- issue one tile's DMA (4 x b128 per thread => 4 ASYNCcnt per wave) ----
  auto issue = [&](int kt, int buf) {
    async_b128(lds_off(&ldsA[buf][r0 * LDT + kc]), Abase + (long)r0 * K + kt);
    async_b128(lds_off(&ldsA[buf][r1 * LDT + kc]), Abase + (long)r1 * K + kt);
    async_b128(lds_off(&ldsB[buf][r0 * LDT + kc]), Wbase + (long)r0 * K + kt);
    async_b128(lds_off(&ldsB[buf][r1 * LDT + kc]), Wbase + (long)r1 * K + kt);
  };

  int nk = K / BK;
  issue(0, 0);
  for (int t = 0; t < nk; ++t) {
    int cur = t & 1;
    if (t + 1 < nk) {
      issue((t + 1) * BK, cur ^ 1);
      // current tile's 4 loads complete in-order while next tile's 4 stay in flight
      asm volatile("s_wait_asynccnt 0x4" ::: "memory");
    } else {
      asm volatile("s_wait_asynccnt 0x0" ::: "memory");
    }
    __syncthreads();

    // A fragment (16x32): lane l(+16*hi): row M=l, K = hi*8+e (e<8), 16+hi*8+(e-8)
    v16bf afrag[4], bfrag[2];
#pragma unroll
    for (int i = 0; i < 4; ++i) {
      int row = wm + i * 16 + l;
      ((v8bf*)&afrag[i])[0] = *(const v8bf*)&ldsA[cur][row * LDT + hi * 8];
      ((v8bf*)&afrag[i])[1] = *(const v8bf*)&ldsA[cur][row * LDT + 16 + hi * 8];
    }
    // B fragment (32x16): lane l(+16*hi): col N=l, K = hi*8+e (e<8), 16+hi*8+(e-8)
#pragma unroll
    for (int j = 0; j < 2; ++j) {
      int n = wn + j * 16 + l;
      ((v8bf*)&bfrag[j])[0] = *(const v8bf*)&ldsB[cur][n * LDT + hi * 8];
      ((v8bf*)&bfrag[j])[1] = *(const v8bf*)&ldsB[cur][n * LDT + 16 + hi * 8];
    }
#pragma unroll
    for (int i = 0; i < 4; ++i)
#pragma unroll
      for (int j = 0; j < 2; ++j)
        acc[i][j] = __builtin_amdgcn_wmma_f32_16x16x32_bf16(
            false, afrag[i], false, bfrag[j], (short)0, acc[i][j], false, false);
    __syncthreads();   // all waves done reading before this buffer is DMA-overwritten
  }

  // epilogue: elem t of C/D frag: M = frag_m0 + hi*8 + t, N = frag_n0 + l
#pragma unroll
  for (int i = 0; i < 4; ++i) {
    long gm = m0 + wm + i * 16 + hi * 8;
#pragma unroll
    for (int j = 0; j < 2; ++j) {
      long gn = n0 + wn + j * 16 + l;
      float bs = bias[gn];
      float* cp = C + gm * N + gn;
      const float* rp = RES ? (resid + gm * N + gn) : nullptr;
#pragma unroll
      for (int t = 0; t < 8; ++t) {
        float v = acc[i][j][t] + bs;
        if (RES) v += rp[(long)t * N];
        cp[(long)t * N] = v;
      }
    }
  }
}

// ---------------- host launch ----------------
extern "C" void kernel_launch(void* const* d_in, const int* in_sizes, int n_in,
                              void* d_out, int out_size, void* d_ws, size_t ws_size,
                              hipStream_t stream) {
  const float* hidden = (const float*)d_in[0];
  const unsigned char* amask = (const unsigned char*)d_in[1];
  const float* qkv_w = (const float*)d_in[2];
  const float* qkv_b = (const float*)d_in[3];
  const float* out_w = (const float*)d_in[4];
  const float* out_b = (const float*)d_in[5];
  /* d_in[6] query_weights: unused by the reference */
  const float* key_w = (const float*)d_in[7];
  const float* n1g = (const float*)d_in[8];
  const float* n1b = (const float*)d_in[9];
  const float* n2g = (const float*)d_in[10];
  const float* n2b = (const float*)d_in[11];
  const float* l1w = (const float*)d_in[12];
  const float* l1b = (const float*)d_in[13];
  const float* l2w = (const float*)d_in[14];
  const float* l2b = (const float*)d_in[15];
  float* outp = (float*)d_out;
  (void)in_sizes; (void)n_in; (void)out_size; (void)ws_size;

  char* ws = (char*)d_ws;
  size_t off = 0;
  auto alloc = [&](size_t bytes) {
    char* p = ws + off;
    off += (bytes + 255) & ~(size_t)255;
    return p;
  };
  unsigned short* wkv   = (unsigned short*)alloc(2048ull * 1024 * 2);  //  4 MB
  unsigned short* wout  = (unsigned short*)alloc(1024ull * 1024 * 2);  //  2 MB
  unsigned short* w1    = (unsigned short*)alloc(4096ull * 1024 * 2);  //  8 MB
  unsigned short* w2    = (unsigned short*)alloc(1024ull * 2048 * 2);  //  4 MB
  unsigned short* actb  = (unsigned short*)alloc((size_t)MTOT * 1024 * 2); // 16 MB (h1 / attn / h2)
  float* bigf           = (float*)alloc((size_t)MTOT * 4096 * 4);      // 128 MB (kv then ff)
  float* hid2           = (float*)alloc((size_t)MTOT * 1024 * 4);      // 32 MB
  unsigned short* gated = (unsigned short*)alloc((size_t)MTOT * 2048 * 2); // 32 MB

  // weight conversion (q third of qkv_w is dead — skip it)
  cvt_bf16<<<512, 256, 0, stream>>>(qkv_w + 1024ull * 1024, wkv, 2048ll * 1024);
  cvt_bf16<<<512, 256, 0, stream>>>(out_w, wout, 1024ll * 1024);
  cvt_bf16<<<512, 256, 0, stream>>>(l1w, w1, 4096ll * 1024);
  cvt_bf16<<<512, 256, 0, stream>>>(l2w, w2, 1024ll * 2048);

  // LN1 -> bf16 activations
  ln_bf16<<<MTOT, 256, 0, stream>>>(hidden, n1g, n1b, actb);
  // KV projection: [8192,1024] x [1024,2048] (+ qkv_b[1024:3072])
  wmma_gemm_bf16<false><<<dim3(2048 / BN, MTOT / BM), 256, 0, stream>>>(
      actb, wkv, qkv_b + 1024, nullptr, bigf, MTOT, 2048, 1024);
  // global-key attention -> bf16 [8192,1024]
  fast_attn<<<128, 256, 0, stream>>>(bigf, key_w, amask, actb);
  // output projection + residual(hidden)
  wmma_gemm_bf16<true><<<dim3(1024 / BN, MTOT / BM), 256, 0, stream>>>(
      actb, wout, out_b, hidden, hid2, MTOT, 1024, 1024);
  // LN2 -> bf16
  ln_bf16<<<MTOT, 256, 0, stream>>>(hid2, n2g, n2b, actb);
  // FF up: [8192,1024] x [1024,4096]
  wmma_gemm_bf16<false><<<dim3(4096 / BN, MTOT / BM), 256, 0, stream>>>(
      actb, w1, l1b, nullptr, bigf, MTOT, 4096, 1024);
  // gated ReLU -> bf16 [8192,2048]
  gate_bf16<<<2048, 256, 0, stream>>>(bigf, gated);
  // FF down + residual(hid2) -> final output
  wmma_gemm_bf16<true><<<dim3(1024 / BN, MTOT / BM), 256, 0, stream>>>(
      gated, w2, l2b, hid2, outp, MTOT, 1024, 2048);
}
